// MambaEncoder_87290915324072
// MI455X (gfx1250) — matrix operand order repaired and checked by reading, not compiled, not run on hardware
//
#include <hip/hip_runtime.h>
#include <hip/hip_bf16.h>

// ---------------------------------------------------------------------------
// MI455X (gfx1250) fused Mamba encoder + cross-attention.
// Main GEMMs: LDS-tiled (128x128 block tile, double-buffered) with
// v_wmma_f32_16x16x32_bf16; 64x32 macro-tile (4x2 WMMA) per wave32.
// ---------------------------------------------------------------------------

typedef __attribute__((ext_vector_type(16))) __bf16 v16bf;
typedef __attribute__((ext_vector_type(8)))  float  v8f;

#define N_TOK 4096
#define DM    1024
#define DI    2048
#define DS    16
#define DTR   64
#define DBCW  96     // DT_RANK + 2*D_STATE
#define DH    512

#define MBLK  128
#define NBLK  128
#define LDST  40     // padded LDS row stride (elements); 80 B, 16 B aligned

__device__ __forceinline__ unsigned short f2bf(float f) {
  unsigned int u = __float_as_uint(f);
  u += 0x7FFFu + ((u >> 16) & 1u);          // round-to-nearest-even
  return (unsigned short)(u >> 16);
}
__device__ __forceinline__ float bf2f(unsigned short h) {
  return __uint_as_float(((unsigned int)h) << 16);
}

union FragBF {
  v16bf v;
  uint4 q[2];
};

// ---------------------------------------------------------------------------
// Shared fragment layout (v_wmma_f32_16x16x32_bf16, wave32):
// A fragment (16x32): lane L<16 -> row M=L, K={k0..k0+7},{k0+16..k0+23};
//                     lane L>=16 -> row M=L-16, K={k0+8..15},{k0+24..31}.
// B fragment (32x16): lane L<16 -> col N=L, K={k0..k0+7},{k0+16..k0+23};
//                     lane L>=16 -> col N=L-16, K={k0+8..15},{k0+24..31}.
// C/D (16x16 f32, 8 VGPRs): lane L -> N=L&15, VGPR v -> M = v + 8*(L>>4).
// ---------------------------------------------------------------------------

// LDS-tiled GEMM: C[M,N](f32) = A[M,K](bf16 rm) * B[N,K]^T (bf16 rm).
// Block = 256 thr = 8 waves (2 in M x 4 in N) -> 128x128 block tile.
// Per 32-K step: stage 128x32 A + 128x32 B into double-buffered LDS
// (16 KB -> 64 WMMAs = 64 flops/byte of L2 traffic).
// Requires M%128==0, N%128==0, K%32==0.
__global__ void __launch_bounds__(256)
gemm_nt_bf16_lds(const unsigned short* __restrict__ A,
                 const unsigned short* __restrict__ B,
                 float* __restrict__ C,
                 int M, int N, int K)
{
  __shared__ alignas(16) unsigned short ldsA[2][MBLK * LDST];
  __shared__ alignas(16) unsigned short ldsB[2][MBLK * LDST];

  const int tid  = threadIdx.x;
  const int lane = tid & 31;
  const int w    = tid >> 5;
  const int tilesN = N / NBLK;
  const int m_blk = (int)(blockIdx.x / tilesN) * MBLK;
  const int n_blk = (int)(blockIdx.x % tilesN) * NBLK;
  const int mw = (w >> 2) << 6;              // wave row offset: 0 or 64
  const int nw = (w & 3) << 5;               // wave col offset: 0,32,64,96
  const int half = lane >> 4;
  const int r    = lane & 15;

  // cooperative staging: each thread owns 2 x 16B chunks per tile
  const int row0 = tid >> 2;                 // 0..63
  const int cc   = (tid & 3) << 3;           // element offset 0,8,16,24
  const unsigned short* gA0 = A + (size_t)(m_blk + row0) * K + cc;
  const unsigned short* gA1 = A + (size_t)(m_blk + row0 + 64) * K + cc;
  const unsigned short* gB0 = B + (size_t)(n_blk + row0) * K + cc;
  const unsigned short* gB1 = B + (size_t)(n_blk + row0 + 64) * K + cc;
  const int l0 = row0 * LDST + cc;
  const int l1 = (row0 + 64) * LDST + cc;

  v8f acc[8];                                // acc[mi*2+ni]
  #pragma unroll
  for (int i = 0; i < 8; ++i)
    #pragma unroll
    for (int j = 0; j < 8; ++j) acc[i][j] = 0.0f;

  // prologue: stage k0 = 0 into buffer 0
  uint4 ra0 = *(const uint4*)(gA0);
  uint4 ra1 = *(const uint4*)(gA1);
  uint4 rb0 = *(const uint4*)(gB0);
  uint4 rb1 = *(const uint4*)(gB1);
  *(uint4*)&ldsA[0][l0] = ra0;
  *(uint4*)&ldsA[0][l1] = ra1;
  *(uint4*)&ldsB[0][l0] = rb0;
  *(uint4*)&ldsB[0][l1] = rb1;
  __syncthreads();

  int cur = 0;
  for (int k0 = 0; k0 < K; k0 += 32) {
    const bool more = (k0 + 32) < K;
    if (more) {                               // issue next-step global loads
      ra0 = *(const uint4*)(gA0 + k0 + 32);
      ra1 = *(const uint4*)(gA1 + k0 + 32);
      rb0 = *(const uint4*)(gB0 + k0 + 32);
      rb1 = *(const uint4*)(gB1 + k0 + 32);
    }

    FragBF fa[4], fb[2];                      // fragment reads from LDS
    #pragma unroll
    for (int mi = 0; mi < 4; ++mi) {
      const int row = mw + mi * 16 + r;
      fa[mi].q[0] = *(const uint4*)&ldsA[cur][row * LDST + (half << 3)];
      fa[mi].q[1] = *(const uint4*)&ldsA[cur][row * LDST + 16 + (half << 3)];
    }
    #pragma unroll
    for (int ni = 0; ni < 2; ++ni) {
      const int row = nw + ni * 16 + r;
      fb[ni].q[0] = *(const uint4*)&ldsB[cur][row * LDST + (half << 3)];
      fb[ni].q[1] = *(const uint4*)&ldsB[cur][row * LDST + 16 + (half << 3)];
    }
    #pragma unroll
    for (int mi = 0; mi < 4; ++mi)
      #pragma unroll
      for (int ni = 0; ni < 2; ++ni)
        acc[mi * 2 + ni] = __builtin_amdgcn_wmma_f32_16x16x32_bf16(
            false, fa[mi].v, false, fb[ni].v,
            (short)0, acc[mi * 2 + ni], false, false);

    if (more) {                               // fill the other buffer
      *(uint4*)&ldsA[cur ^ 1][l0] = ra0;
      *(uint4*)&ldsA[cur ^ 1][l1] = ra1;
      *(uint4*)&ldsB[cur ^ 1][l0] = rb0;
      *(uint4*)&ldsB[cur ^ 1][l1] = rb1;
    }
    __syncthreads();
    cur ^= 1;
  }

  #pragma unroll
  for (int mi = 0; mi < 4; ++mi) {
    #pragma unroll
    for (int ni = 0; ni < 2; ++ni) {
      float* pc = C + (size_t)(m_blk + mw + mi * 16 + (half << 3)) * N
                    + n_blk + nw + ni * 16 + r;
      #pragma unroll
      for (int v = 0; v < 8; ++v) pc[(size_t)v * N] = acc[mi * 2 + ni][v];
    }
  }
}

// Register-only GEMM (64x32 macro-tile per wave) for small-N shapes
// (here: N=96 for the x_proj GEMM).  M%64==0, N%32==0, K%32==0.
__global__ void __launch_bounds__(256)
gemm_nt_bf16_reg(const unsigned short* __restrict__ A,
                 const unsigned short* __restrict__ B,
                 float* __restrict__ C,
                 int M, int N, int K)
{
  const int lane = threadIdx.x & 31;
  const int wave = (blockIdx.x << 3) + (threadIdx.x >> 5);
  const int tilesN = N >> 5;
  const int tilesM = M >> 6;
  if (wave >= tilesM * tilesN) return;       // wave-granular: EXEC all-1s
  const int m0 = (wave / tilesN) << 6;
  const int n0 = (wave % tilesN) << 5;
  const int half = lane >> 4;
  const int r    = lane & 15;

  const unsigned short* pa = A + (size_t)(m0 + r) * K + (half << 3);
  const unsigned short* pb = B + (size_t)(n0 + r) * K + (half << 3);
  const size_t strA = (size_t)16 * K;
  const size_t strB = (size_t)16 * K;

  v8f acc[8];
  #pragma unroll
  for (int i = 0; i < 8; ++i)
    #pragma unroll
    for (int j = 0; j < 8; ++j) acc[i][j] = 0.0f;

  for (int k = 0; k < K; k += 32) {
    FragBF fa[4], fb[2];
    #pragma unroll
    for (int mi = 0; mi < 4; ++mi) {
      fa[mi].q[0] = *(const uint4*)(pa + mi * strA + k);
      fa[mi].q[1] = *(const uint4*)(pa + mi * strA + k + 16);
    }
    #pragma unroll
    for (int ni = 0; ni < 2; ++ni) {
      fb[ni].q[0] = *(const uint4*)(pb + ni * strB + k);
      fb[ni].q[1] = *(const uint4*)(pb + ni * strB + k + 16);
    }
    #pragma unroll
    for (int mi = 0; mi < 4; ++mi)
      #pragma unroll
      for (int ni = 0; ni < 2; ++ni)
        acc[mi * 2 + ni] = __builtin_amdgcn_wmma_f32_16x16x32_bf16(
            false, fa[mi].v, false, fb[ni].v,
            (short)0, acc[mi * 2 + ni], false, false);
  }

  #pragma unroll
  for (int mi = 0; mi < 4; ++mi) {
    #pragma unroll
    for (int ni = 0; ni < 2; ++ni) {
      float* pc = C + (size_t)(m0 + mi * 16 + (half << 3)) * N + n0 + ni * 16 + r;
      #pragma unroll
      for (int v = 0; v < 8; ++v) pc[(size_t)v * N] = acc[mi * 2 + ni][v];
    }
  }
}

// --------------------------- elementwise kernels ---------------------------

__global__ void pack_tokens_bf16(const float* __restrict__ img,
                                 const float* __restrict__ txt,
                                 unsigned short* __restrict__ out) {
  int idx = blockIdx.x * blockDim.x + threadIdx.x;
  if (idx >= N_TOK * DM) return;
  int i = idx >> 10, j = idx & 1023;
  float v = (j < DH) ? img[(size_t)i * DH + j] : txt[(size_t)i * DH + (j - DH)];
  out[idx] = f2bf(v);
}

__global__ void cvt_f32_bf16(const float* __restrict__ in,
                             unsigned short* __restrict__ out, int n) {
  int idx = blockIdx.x * blockDim.x + threadIdx.x;
  if (idx < n) out[idx] = f2bf(in[idx]);
}

// text (4096x512) -> text^T as bf16 (512 rows x 4096 K) for att @ text.
__global__ void transpose_text_bf16(const float* __restrict__ txt,
                                    unsigned short* __restrict__ out) {
  int idx = blockIdx.x * blockDim.x + threadIdx.x;
  if (idx >= DH * N_TOK) return;
  int nrow = idx >> 12;          // 0..511
  int k    = idx & 4095;         // 0..4095
  out[idx] = f2bf(txt[(size_t)k * DH + nrow]);
}

// causal depthwise conv (width 4) + bias + SiLU on x = xz[:, :DI]
__global__ void conv_silu(const float* __restrict__ xz,
                          const float* __restrict__ cw,
                          const float* __restrict__ cb,
                          unsigned short* __restrict__ xconv) {
  int idx = blockIdx.x * blockDim.x + threadIdx.x;
  if (idx >= N_TOK * DI) return;
  int t = idx >> 11, c = idx & (DI - 1);
  float s = cb[c];
  #pragma unroll
  for (int k = 0; k < 4; ++k) {
    int tt = t - 3 + k;
    if (tt >= 0) s += xz[(size_t)tt * (2 * DI) + c] * cw[c * 4 + k];
  }
  s = s / (1.0f + __expf(-s));   // SiLU
  xconv[idx] = f2bf(s);
}

__global__ void extract_dtraw(const float* __restrict__ dbc,
                              unsigned short* __restrict__ out) {
  int idx = blockIdx.x * blockDim.x + threadIdx.x;
  if (idx >= N_TOK * DTR) return;
  int t = idx >> 6, j = idx & 63;
  out[idx] = f2bf(dbc[(size_t)t * DBCW + j]);
}

__global__ void dt_softplus(float* __restrict__ dt,
                            const float* __restrict__ bias) {
  int idx = blockIdx.x * blockDim.x + threadIdx.x;
  if (idx >= N_TOK * DI) return;
  int c = idx & (DI - 1);
  float v = dt[idx] + bias[c];
  dt[idx] = (v > 20.0f) ? v : log1pf(__expf(v));
}

// ------------------------------- SSM scan ----------------------------------
// 16 lanes per channel (one per state), 2 channels per wave32, 16 per block.
__global__ void __launch_bounds__(256)
ssm_scan(const float* __restrict__ dt, const unsigned short* __restrict__ xconv,
         const float* __restrict__ dbc, const float* __restrict__ A_log,
         float* __restrict__ yscan) {
  int lane = threadIdx.x & 31;
  int wid  = threadIdx.x >> 5;
  int s = lane & 15;
  int c = (blockIdx.x << 4) + (wid << 1) + (lane >> 4);
  float A = -__expf(A_log[(size_t)c * DS + s]);
  float h = 0.0f;
  for (int t = 0; t < N_TOK; ++t) {
    float dtv = dt[(size_t)t * DI + c];
    float xv  = bf2f(xconv[(size_t)t * DI + c]);
    float Bv  = dbc[(size_t)t * DBCW + DTR + s];
    float Cv  = dbc[(size_t)t * DBCW + DTR + DS + s];
    h = __expf(dtv * A) * h + (dtv * xv) * Bv;
    float y = h * Cv;
    y += __shfl_xor(y, 8, 32);   // reduce within the 16-lane state group
    y += __shfl_xor(y, 4, 32);
    y += __shfl_xor(y, 2, 32);
    y += __shfl_xor(y, 1, 32);
    if (s == 0) yscan[(size_t)t * DI + c] = y;
  }
}

__global__ void gate_y(const float* __restrict__ yscan,
                       const unsigned short* __restrict__ xconv,
                       const float* __restrict__ Dp,
                       const float* __restrict__ xz,
                       unsigned short* __restrict__ ybf) {
  int idx = blockIdx.x * blockDim.x + threadIdx.x;
  if (idx >= N_TOK * DI) return;
  int t = idx >> 11, c = idx & (DI - 1);
  float yv = yscan[idx] + Dp[c] * bf2f(xconv[idx]);
  float z  = xz[(size_t)t * (2 * DI) + DI + c];
  float sz = z / (1.0f + __expf(-z));
  ybf[idx] = f2bf(yv * sz);
}

// ------------------------------ softmax ------------------------------------
__global__ void __launch_bounds__(256)
softmax_row_bf16(const float* __restrict__ scores,
                 unsigned short* __restrict__ att) {
  __shared__ float red[256];
  int i = blockIdx.x, tid = threadIdx.x;
  const float* row = scores + (size_t)i * N_TOK;
  float vals[16];
  float lmax = -3.402823466e38f;
  #pragma unroll
  for (int k = 0; k < 16; ++k) {
    vals[k] = row[tid + (k << 8)];
    lmax = fmaxf(lmax, vals[k]);
  }
  red[tid] = lmax; __syncthreads();
  for (int o = 128; o > 0; o >>= 1) {
    if (tid < o) red[tid] = fmaxf(red[tid], red[tid + o]);
    __syncthreads();
  }
  float m = red[0]; __syncthreads();
  float lsum = 0.0f;
  #pragma unroll
  for (int k = 0; k < 16; ++k) { vals[k] = __expf(vals[k] - m); lsum += vals[k]; }
  red[tid] = lsum; __syncthreads();
  for (int o = 128; o > 0; o >>= 1) {
    if (tid < o) red[tid] += red[tid + o];
    __syncthreads();
  }
  float inv = 1.0f / red[0];
  unsigned short* orow = att + (size_t)i * N_TOK;
  #pragma unroll
  for (int k = 0; k < 16; ++k) orow[tid + (k << 8)] = f2bf(vals[k] * inv);
}

// ------------------- final residual + L2 norm + split ----------------------
__global__ void __launch_bounds__(256)
fuse_norm_out(const float* __restrict__ mbuf, const float* __restrict__ cam_out,
              const float* __restrict__ img, const float* __restrict__ alpha,
              float* __restrict__ out) {
  __shared__ float red[256];
  int i = blockIdx.x, tid = threadIdx.x;
  float a = alpha[0];
  float vals[4];
  float ss = 0.0f;
  #pragma unroll
  for (int k = 0; k < 4; ++k) {
    int j = tid + (k << 8);
    float cam = (j < DH) ? a * cam_out[(size_t)i * DH + j]
                         : img[(size_t)i * DH + (j - DH)];
    float r = mbuf[(size_t)i * DM + j] + cam;
    vals[k] = r; ss += r * r;
  }
  red[tid] = ss; __syncthreads();
  for (int o = 128; o > 0; o >>= 1) {
    if (tid < o) red[tid] += red[tid + o];
    __syncthreads();
  }
  float scale = 1.0f / fmaxf(sqrtf(red[0]), 1e-12f);
  #pragma unroll
  for (int k = 0; k < 4; ++k) {
    int j = tid + (k << 8);
    float r = vals[k] * scale;
    if (j < DH) out[(size_t)i * DH + j] = r;                                  // tuple[0]
    else        out[(size_t)N_TOK * DH + (size_t)i * DH + (j - DH)] = r;      // tuple[1]
  }
}

// ---------------------------------------------------------------------------
extern "C" void kernel_launch(void* const* d_in, const int* in_sizes, int n_in,
                              void* d_out, int out_size, void* d_ws, size_t ws_size,
                              hipStream_t stream) {
  const float* img   = (const float*)d_in[0];
  const float* txt   = (const float*)d_in[1];
  const float* w_in  = (const float*)d_in[2];
  const float* cw    = (const float*)d_in[3];
  const float* cb    = (const float*)d_in[4];
  const float* w_xp  = (const float*)d_in[5];
  const float* w_dt  = (const float*)d_in[6];
  const float* b_dt  = (const float*)d_in[7];
  const float* A_log = (const float*)d_in[8];
  const float* Dp    = (const float*)d_in[9];
  const float* w_out = (const float*)d_in[10];
  const float* alpha = (const float*)d_in[11];
  float* out = (float*)d_out;

  char* p = (char*)d_ws;
  auto alloc = [&](size_t bytes) -> void* {
    void* r = (void*)p;
    p += (bytes + 255) & ~(size_t)255;
    return r;
  };
  unsigned short* tok_bf   = (unsigned short*)alloc((size_t)N_TOK * DM * 2);
  unsigned short* win_bf   = (unsigned short*)alloc((size_t)2 * DI * DM * 2);
  float*          xz       = (float*)alloc((size_t)N_TOK * 2 * DI * 4);
  unsigned short* xconv_bf = (unsigned short*)alloc((size_t)N_TOK * DI * 2);
  unsigned short* wxp_bf   = (unsigned short*)alloc((size_t)DBCW * DI * 2);
  float*          dbc      = (float*)alloc((size_t)N_TOK * DBCW * 4);
  unsigned short* dtraw_bf = (unsigned short*)alloc((size_t)N_TOK * DTR * 2);
  unsigned short* wdt_bf   = (unsigned short*)alloc((size_t)DI * DTR * 2);
  float*          dtbuf    = (float*)alloc((size_t)N_TOK * DI * 4);
  float*          yscan    = (float*)alloc((size_t)N_TOK * DI * 4);
  unsigned short* y_bf     = (unsigned short*)alloc((size_t)N_TOK * DI * 2);
  unsigned short* wout_bf  = (unsigned short*)alloc((size_t)DM * DI * 2);
  float*          mbuf     = (float*)alloc((size_t)N_TOK * DM * 4);
  unsigned short* img_bf   = (unsigned short*)alloc((size_t)N_TOK * DH * 2);
  unsigned short* txt_bf   = (unsigned short*)alloc((size_t)N_TOK * DH * 2);
  unsigned short* txtT_bf  = (unsigned short*)alloc((size_t)DH * N_TOK * 2);
  float*          scores   = (float*)alloc((size_t)N_TOK * N_TOK * 4);
  unsigned short* att_bf   = (unsigned short*)alloc((size_t)N_TOK * N_TOK * 2);
  float*          cam_out  = (float*)alloc((size_t)N_TOK * DH * 4);

  auto grid1 = [](size_t n) { return (unsigned)((n + 255) / 256); };
  auto gridLds = [](int M, int N) {
    return (unsigned)((M / MBLK) * (N / NBLK));   // one 128x128 tile per block
  };
  auto gridReg = [](int M, int N) {
    int tiles = (M / 64) * (N / 32);
    return (unsigned)((tiles + 7) / 8);
  };

  // bf16 conversions / packing
  pack_tokens_bf16<<<grid1((size_t)N_TOK * DM), 256, 0, stream>>>(img, txt, tok_bf);
  cvt_f32_bf16<<<grid1((size_t)2 * DI * DM), 256, 0, stream>>>(w_in, win_bf, 2 * DI * DM);
  cvt_f32_bf16<<<grid1((size_t)DBCW * DI), 256, 0, stream>>>(w_xp, wxp_bf, DBCW * DI);
  cvt_f32_bf16<<<grid1((size_t)DI * DTR), 256, 0, stream>>>(w_dt, wdt_bf, DI * DTR);
  cvt_f32_bf16<<<grid1((size_t)DM * DI), 256, 0, stream>>>(w_out, wout_bf, DM * DI);
  cvt_f32_bf16<<<grid1((size_t)N_TOK * DH), 256, 0, stream>>>(img, img_bf, N_TOK * DH);
  cvt_f32_bf16<<<grid1((size_t)N_TOK * DH), 256, 0, stream>>>(txt, txt_bf, N_TOK * DH);
  transpose_text_bf16<<<grid1((size_t)DH * N_TOK), 256, 0, stream>>>(txt, txtT_bf);

  // --- Mamba chain ---
  gemm_nt_bf16_lds<<<gridLds(N_TOK, 2 * DI), 256, 0, stream>>>(tok_bf, win_bf, xz,
                                                               N_TOK, 2 * DI, DM);
  conv_silu<<<grid1((size_t)N_TOK * DI), 256, 0, stream>>>(xz, cw, cb, xconv_bf);
  gemm_nt_bf16_reg<<<gridReg(N_TOK, DBCW), 256, 0, stream>>>(xconv_bf, wxp_bf, dbc,
                                                             N_TOK, DBCW, DI);
  extract_dtraw<<<grid1((size_t)N_TOK * DTR), 256, 0, stream>>>(dbc, dtraw_bf);
  gemm_nt_bf16_lds<<<gridLds(N_TOK, DI), 256, 0, stream>>>(dtraw_bf, wdt_bf, dtbuf,
                                                           N_TOK, DI, DTR);
  dt_softplus<<<grid1((size_t)N_TOK * DI), 256, 0, stream>>>(dtbuf, b_dt);
  ssm_scan<<<DI / 16, 256, 0, stream>>>(dtbuf, xconv_bf, dbc, A_log, yscan);
  gate_y<<<grid1((size_t)N_TOK * DI), 256, 0, stream>>>(yscan, xconv_bf, Dp, xz, y_bf);
  gemm_nt_bf16_lds<<<gridLds(N_TOK, DM), 256, 0, stream>>>(y_bf, wout_bf, mbuf,
                                                           N_TOK, DM, DI);

  // --- cross-attention branch ---
  gemm_nt_bf16_lds<<<gridLds(N_TOK, N_TOK), 256, 0, stream>>>(img_bf, txt_bf, scores,
                                                              N_TOK, N_TOK, DH);
  softmax_row_bf16<<<N_TOK, 256, 0, stream>>>(scores, att_bf);
  gemm_nt_bf16_lds<<<gridLds(N_TOK, DH), 256, 0, stream>>>(att_bf, txtT_bf, cam_out,
                                                           N_TOK, DH, N_TOK);

  // --- fuse, normalize, split-write ---
  fuse_norm_out<<<N_TOK, 256, 0, stream>>>(mbuf, cam_out, img, alpha, out);
}
